// RNN_88356067213395
// MI455X (gfx1250) — matrix operand order repaired and checked by reading, not compiled, not run on hardware
//
#include <hip/hip_runtime.h>
#include <math.h>

// 2-layer tanh RNN scan, B=4096, T=2048, H=32, future=0.
// One wave owns a 16-row batch tile for all T steps; weights + both hidden
// states live in registers the whole time. State kept transposed (G = h^T,
// 32x16) so recurrent GEMMs are G_new = W @ G (A = weight rows, B = state).
//
// Hidden-index trick: the K-ordering of all weight A-fragments is AMD's table order
//   k = 8h + s (s < 8), k = 16 + 8h + (s - 8) (s >= 8), h = lane / 16
// which makes the D(C-layout) -> B-fragment conversion completely LANE-LOCAL:
//   B[i] = f16(tanh(d_lo[i])), B[i+8] = f16(tanh(d_hi[i]))   (no shuffles!)
// The readout h1.w_out is a WMMA too, so out[b] lands directly in lane b.
// Layer-1's two GEMMs are issued as independent WMMAs (+ v8f add) so their
// XDL latencies overlap instead of chaining.

typedef __attribute__((ext_vector_type(16))) _Float16 v16h;
typedef __attribute__((ext_vector_type(8)))  float    v8f;
typedef __attribute__((ext_vector_type(4)))  float    v4f;

#define RNN_B 4096
#define RNN_T 2048
#define RNN_H 32

static __device__ __forceinline__ float fast_tanh(float v) {
#if __has_builtin(__builtin_amdgcn_tanhf)
  return __builtin_amdgcn_tanhf(v);  // v_tanh_f32 (TRANS, co-executes with WMMA)
#elif __has_builtin(__builtin_amdgcn_tanh_f32)
  return __builtin_amdgcn_tanh_f32(v);
#else
  return tanhf(v);
#endif
}

static __device__ __forceinline__ v8f wmma_f16(v16h a, v16h b, v8f c) {
  return __builtin_amdgcn_wmma_f32_16x16x32_f16(false, a, false, b, (short)0, c,
                                                false, false);
}

static __device__ __forceinline__ int tau(int k) {
  return (k >= 8 && k < 24) ? (k ^ 24) : k;
}

__global__ __launch_bounds__(256) void rnn_scan_kernel(
    const float* __restrict__ x,
    const float* __restrict__ w_ih0, const float* __restrict__ w_hh0,
    const float* __restrict__ b_ih0, const float* __restrict__ b_hh0,
    const float* __restrict__ w_ih1, const float* __restrict__ w_hh1,
    const float* __restrict__ b_ih1, const float* __restrict__ b_hh1,
    const float* __restrict__ w_out, const float* __restrict__ b_out,
    float* __restrict__ out) {
  const int lane = threadIdx.x & 31;
  const int wave = threadIdx.x >> 5;
  const int tile = blockIdx.x * (blockDim.x >> 5) + wave;  // 16 batch rows
  const int b0   = tile * 16;
  if (b0 >= RNN_B) return;  // wave-uniform; EXEC stays all-ones for WMMA

  const int  m      = lane & 15;       // A row / batch column within tile
  const bool hiHalf = lane >= 16;
  const int  kbase  = hiHalf ? 8 : 0;  // A-fragment K-position base

  // ---- Weight rows as WMMA A-fragments, AMD table K-order ----
  auto loadA = [&](const float* Wp, int rowOff) -> v16h {
    v16h a;
    const int r = rowOff + m;
#pragma unroll
    for (int j = 0; j < 8; ++j) {
      const int kpos = kbase + ((j < 4) ? 2 * j : 16 + 2 * (j - 4));
      const int k    = kpos;
      a[2 * j]     = (_Float16)Wp[r * RNN_H + k];
      a[2 * j + 1] = (_Float16)Wp[r * RNN_H + k + 1];
    }
    return a;
  };
  const v16h A_hh0_lo = loadA(w_hh0, 0), A_hh0_hi = loadA(w_hh0, 16);
  const v16h A_ih1_lo = loadA(w_ih1, 0), A_ih1_hi = loadA(w_ih1, 16);
  const v16h A_hh1_lo = loadA(w_hh1, 0), A_hh1_hi = loadA(w_hh1, 16);

  // Readout fragment: row 0 = w_out (table K-order), other rows zero.
  v16h A_out = {};
  if (m == 0) {
#pragma unroll
    for (int j = 0; j < 8; ++j) {
      const int kpos = kbase + ((j < 4) ? 2 * j : 16 + 2 * (j - 4));
      const int k    = kpos;
      A_out[2 * j]     = (_Float16)w_out[k];
      A_out[2 * j + 1] = (_Float16)w_out[k + 1];
    }
  }

  // ---- Per-lane C-layout broadcast vectors (row n = i + 8*hiHalf) ----
  v8f bias0_lo, bias0_hi, bias1_lo, bias1_hi, wi0_lo, wi0_hi;
#pragma unroll
  for (int i = 0; i < 8; ++i) {
    const int nlo = i + (hiHalf ? 8 : 0);
    const int nhi = 16 + nlo;
    bias0_lo[i] = b_ih0[nlo] + b_hh0[nlo];
    bias0_hi[i] = b_ih0[nhi] + b_hh0[nhi];
    bias1_lo[i] = b_ih1[nlo] + b_hh1[nlo];
    bias1_hi[i] = b_ih1[nhi] + b_hh1[nhi];
    wi0_lo[i] = w_ih0[nlo];  // (H,1)
    wi0_hi[i] = w_ih0[nhi];
  }
  v8f c_out = {};
  c_out[0] = b_out[0];  // M=0 row gets the output bias
  const v8f zeroC = {};

  // ---- Recurrent scan (quad-batched along t for b128 I/O) ----
  v16h G0 = {};  // h0^T fragment (table K-order), zeros
  v16h G1 = {};  // h1^T fragment
  const float* xlane   = x + (size_t)(b0 + m) * RNN_T;
  float*       outlane = out + (size_t)(b0 + m) * RNN_T;

  v4f xq = __builtin_nontemporal_load((const v4f*)xlane);  // t = 0..3
  for (int tq = 0; tq < RNN_T; tq += 4) {
    // Branchless prefetch: wraps to t=0 on the last quad (loaded, unused).
    const int tn = (tq + 4) & (RNN_T - 1);
    const v4f xq_next = __builtin_nontemporal_load((const v4f*)(xlane + tn));

    v4f oq;
#pragma unroll
    for (int u = 0; u < 4; ++u) {
      const float xv = xq[u];

      // Layer 0: C = x*w_ih0 + (b_ih0 + b_hh0); D = W_hh0 @ G0 + C
      v8f c0lo, c0hi;
#pragma unroll
      for (int i = 0; i < 8; ++i) {
        c0lo[i] = fmaf(xv, wi0_lo[i], bias0_lo[i]);
        c0hi[i] = fmaf(xv, wi0_hi[i], bias0_hi[i]);
      }
      const v8f d0lo = wmma_f16(A_hh0_lo, G0, c0lo);
      const v8f d0hi = wmma_f16(A_hh0_hi, G0, c0hi);

      // tanh + lane-local repack into next B fragment (table K-order)
#pragma unroll
      for (int i = 0; i < 8; ++i) {
        G0[i]     = (_Float16)fast_tanh(d0lo[i]);
        G0[i + 8] = (_Float16)fast_tanh(d0hi[i]);
      }

      // Layer 1: two INDEPENDENT WMMAs (latencies overlap), then v8f add
      const v8f p1lo = wmma_f16(A_ih1_lo, G0, bias1_lo);
      const v8f p1hi = wmma_f16(A_ih1_hi, G0, bias1_hi);
      const v8f q1lo = wmma_f16(A_hh1_lo, G1, zeroC);
      const v8f q1hi = wmma_f16(A_hh1_hi, G1, zeroC);
      const v8f d1lo = p1lo + q1lo;
      const v8f d1hi = p1hi + q1hi;

#pragma unroll
      for (int i = 0; i < 8; ++i) {
        G1[i]     = (_Float16)fast_tanh(d1lo[i]);
        G1[i + 8] = (_Float16)fast_tanh(d1hi[i]);
      }

      // Readout as a WMMA: row 0 of D = w_out . h1 + b_out, lands in lane b
      const v8f dout = wmma_f16(A_out, G1, c_out);
      oq[u] = dout[0];
    }
    if (!hiHalf) {
      __builtin_nontemporal_store(oq, (v4f*)(outlane + tq));
    }
    xq = xq_next;
  }
}

extern "C" void kernel_launch(void* const* d_in, const int* in_sizes, int n_in,
                              void* d_out, int out_size, void* d_ws,
                              size_t ws_size, hipStream_t stream) {
  (void)in_sizes; (void)n_in; (void)d_ws; (void)ws_size; (void)out_size;
  const float* xp     = (const float*)d_in[0];
  const float* w_ih0  = (const float*)d_in[1];
  const float* w_hh0  = (const float*)d_in[2];
  const float* b_ih0  = (const float*)d_in[3];
  const float* b_hh0  = (const float*)d_in[4];
  const float* w_ih1  = (const float*)d_in[5];
  const float* w_hh1  = (const float*)d_in[6];
  const float* b_ih1  = (const float*)d_in[7];
  const float* b_hh1  = (const float*)d_in[8];
  const float* w_outp = (const float*)d_in[9];
  const float* b_outp = (const float*)d_in[10];
  float* outp = (float*)d_out;

  const int tiles = RNN_B / 16;  // 256 independent waves
  const int wavesPerBlock = 8;   // 256 threads
  dim3 grid(tiles / wavesPerBlock), block(wavesPerBlock * 32);
  rnn_scan_kernel<<<grid, block, 0, stream>>>(
      xp, w_ih0, w_hh0, b_ih0, b_hh0, w_ih1, w_hh1, b_ih1, b_hh1, w_outp,
      b_outp, outp);
}
